// Pulse_extraction_17927193494002
// MI455X (gfx1250) — matrix operand order repaired and checked by reading, not compiled, not run on hardware
//
#include <hip/hip_runtime.h>

typedef __attribute__((ext_vector_type(16))) _Float16 v16h;
typedef __attribute__((ext_vector_type(8)))  _Float16 v8h;
typedef __attribute__((ext_vector_type(8)))  float    v8f;

// Problem constants (reference: T=4, B=16, C=512, H=W=16 -> N=256)
constexpr int TT = 4;
constexpr int BB = 16;
constexpr int CC = 512;
constexpr int NN = 256;
constexpr int SS = BB * CC * NN;     // elements per time step = 2,097,152
constexpr int TBATCH = TT * BB;      // 64 batched GEMMs

// ---------------------------------------------------------------------------
// Async raw copy of 16 bytes global->LDS (CDNA5 GLOBAL_LOAD_ASYNC_TO_LDS_B128,
// tracked by ASYNCcnt). lds_dst/gsrc must be 16B aligned.
// ---------------------------------------------------------------------------
__device__ __forceinline__ void async_copy_b128(_Float16* lds_dst,
                                                const _Float16* gsrc) {
  unsigned lds_off = (unsigned)(unsigned long long)(uintptr_t)lds_dst;
  unsigned long long ga = (unsigned long long)(uintptr_t)gsrc;
  asm volatile("global_load_async_to_lds_b128 %0, %1, off"
               :: "v"(lds_off), "v"(ga) : "memory");
}
__device__ __forceinline__ void wait_async0() {
  asm volatile("s_wait_asynccnt 0" ::: "memory");
}

// ---------------------------------------------------------------------------
// All-f16 LDS-tiled GEMM, double-buffered async staging.
//   block = 256 threads (8 wave32), tile = 64(M) x 128(N), K-chunks of 32.
//   Waves in 2x4 grid, each wave owns 32x32 = 2x2 WMMA accumulators.
// A: row-major [M][K] (lda). Bt: column-major B, i.e. [N][K] (ldbt).
// LDS A rows hold K contiguously so the per-lane A fragment
// (K = hi*16+0..15, matching B) is one
// contiguous 32B v16h read. B fragment (K = hi*16+0..15) is contiguous by
// construction in the [n][k] tile. Staging is pure async b128; one barrier
// per K-iter with ping-pong buffers.
// ---------------------------------------------------------------------------
__device__ __forceinline__ void gemm_f16_64x128(const _Float16* __restrict__ A,
                                                int lda,
                                                const _Float16* __restrict__ Bt,
                                                int ldbt, int m0, int n0, int K,
                                                v8f acc[2][2]) {
  __shared__ alignas(32) _Float16 As[2][64 * 32];    // 2 x 4 KB
  __shared__ alignas(32) _Float16 Bs[2][128 * 32];   // 2 x 8 KB
  const int tid  = threadIdx.x;
  const int wave = tid >> 5;
  const int lane = tid & 31;
  const int hi = lane >> 4, lo = lane & 15;
  const int wm = wave >> 2, wn = wave & 3;

  auto stage = [&](int buf, int kb) {
    // A tile: 64 rows x 32 k, 8 halves per thread, contiguous K-groups.
    const int row  = tid >> 2;                       // 0..63
    const int g    = tid & 3;                        // k-group of 8
    const int posg = g * 8;                          // 0,8,16,24
    async_copy_b128(&As[buf][row * 32 + posg],
                    A + (size_t)(m0 + row) * lda + kb + g * 8);
    // B tile: 128 n-rows x 32 k, 16 halves per thread, contiguous in k.
    const int n  = tid >> 1;                         // 0..127
    const int kk = (tid & 1) * 16;                   // 0 or 16
    const _Float16* bsrc = Bt + (size_t)(n0 + n) * ldbt + kb + kk;
    async_copy_b128(&Bs[buf][n * 32 + kk], bsrc);
    async_copy_b128(&Bs[buf][n * 32 + kk + 8], bsrc + 8);
  };

  stage(0, 0);
  wait_async0();
  __syncthreads();

  int buf = 0;
  for (int kb = 0; kb < K; kb += 32) {
    const bool more = (kb + 32) < K;
    if (more) stage(buf ^ 1, kb + 32);

    v16h af[2], bf[2];
#pragma unroll
    for (int i = 0; i < 2; ++i)
      af[i] = *(const v16h*)&As[buf][(wm * 32 + i * 16 + lo) * 32 + hi * 16];
#pragma unroll
    for (int j = 0; j < 2; ++j)
      bf[j] = *(const v16h*)&Bs[buf][(wn * 32 + j * 16 + lo) * 32 + hi * 16];

#pragma unroll
    for (int i = 0; i < 2; ++i)
#pragma unroll
      for (int j = 0; j < 2; ++j)
        acc[i][j] = __builtin_amdgcn_wmma_f32_16x16x32_f16(
            false, af[i], false, bf[j], (short)0, acc[i][j], false, false);

    if (more) {
      wait_async0();
      __syncthreads();
      buf ^= 1;
    }
  }
}

// ---------------------------------------------------------------------------
// One-time f32 -> f16 weight conversion (512x512 = 262144 elements).
// ---------------------------------------------------------------------------
__global__ void __launch_bounds__(256) cvt_w_f16(const float* __restrict__ src,
                                                 _Float16* __restrict__ dst) {
  const int i = blockIdx.x * 256 + threadIdx.x;
  dst[i] = (_Float16)src[i];
}

// ---------------------------------------------------------------------------
// x [t,b,c,n] f32 -> xT [t,b,n,c] f16 (column-major B operand for qkv GEMMs).
// ---------------------------------------------------------------------------
__global__ void __launch_bounds__(256) transpose_x_f16(
    const float* __restrict__ x, _Float16* __restrict__ xT) {
  const int idx = blockIdx.x * 256 + threadIdx.x;    // over E = TT*SS
  const int tb = idx / (CC * NN);
  const int c  = (idx / NN) % CC;
  const int n  = idx % NN;
  xT[(size_t)tb * NN * CC + (size_t)n * CC + c] = (_Float16)x[idx];
}

// ---------------------------------------------------------------------------
// QKV branch GEMM: U = BN(W @ X). M=512, N=256, K=512. grid=(16, 64).
// W is f16 [C][C]; Xt is f16 [tb][n][c] (column-major).
// ---------------------------------------------------------------------------
__global__ void __launch_bounds__(256) gemm_qkv_bn(
    const _Float16* __restrict__ W, const _Float16* __restrict__ Xt,
    const float* __restrict__ gamma, const float* __restrict__ beta,
    const float* __restrict__ mean, const float* __restrict__ var,
    float* __restrict__ U) {
  const int tb = blockIdx.y;
  const int m0 = (blockIdx.x >> 1) * 64;   // 8 M-blocks
  const int n0 = (blockIdx.x & 1) * 128;   // 2 N-blocks
  v8f acc[2][2] = {};
  gemm_f16_64x128(W, CC, Xt + (size_t)tb * NN * CC, CC, m0, n0, CC, acc);
  const int wave = threadIdx.x >> 5, lane = threadIdx.x & 31;
  const int hi = lane >> 4, lo = lane & 15;
  const int wm = wave >> 2, wn = wave & 3;
  const size_t boff = (size_t)tb * CC * NN;
#pragma unroll
  for (int i = 0; i < 2; ++i)
#pragma unroll
    for (int j = 0; j < 2; ++j) {
      const int n = n0 + wn * 32 + j * 16 + lo;
#pragma unroll
      for (int r = 0; r < 8; ++r) {
        int m = m0 + wm * 32 + i * 16 + hi * 8 + r;
        float inv  = gamma[m] * rsqrtf(var[m] + 1e-5f);
        float bias = beta[m] - mean[m] * inv;
        U[boff + (size_t)m * NN + n] = acc[i][j][r] * inv + bias;
      }
    }
}

// ---------------------------------------------------------------------------
// LIF scan over T for q/k/v. q spikes written [c][n] (A of attn GEMM);
// k and v spikes written transposed [n][c] / [m][c] (A resp. Bt of kv GEMM).
// v-branch spikes also go to the second half of d_out (reference output #2).
// ---------------------------------------------------------------------------
__global__ void __launch_bounds__(256) lif_qkv(
    const float* __restrict__ uq, const float* __restrict__ uk,
    const float* __restrict__ uv, _Float16* __restrict__ sq,
    _Float16* __restrict__ skT, _Float16* __restrict__ svT,
    float* __restrict__ v_out) {
  const int idx = blockIdx.x * blockDim.x + threadIdx.x;  // over SS
  const int b = idx / (CC * NN);
  const int c = (idx / NN) % CC;
  const int n = idx % NN;
  const size_t toff = (size_t)b * NN * CC + (size_t)n * CC + c;
  float vq = 0.f, vk = 0.f, vv = 0.f;
  for (int t = 0; t < TT; ++t) {
    const size_t o = (size_t)t * SS + idx;
    vq += (uq[o] - vq) * 0.5f;
    float s = (vq >= 1.0f) ? 1.0f : 0.0f;
    sq[o] = (_Float16)s;
    vq *= (1.0f - s);

    vk += (uk[o] - vk) * 0.5f;
    s = (vk >= 1.0f) ? 1.0f : 0.0f;
    skT[(size_t)t * SS + toff] = (_Float16)s;
    vk *= (1.0f - s);

    vv += (uv[o] - vv) * 0.5f;
    s = (vv >= 1.0f) ? 1.0f : 0.0f;
    svT[(size_t)t * SS + toff] = (_Float16)s;
    v_out[o] = s;
    vv *= (1.0f - s);
  }
}

// ---------------------------------------------------------------------------
// kv = k^T @ v : M=N=256, K=512. Stored TRANSPOSED as kvT[m][n] f16 so it is
// the column-major B operand of the attention GEMM. grid=(8, 64).
// ---------------------------------------------------------------------------
__global__ void __launch_bounds__(256) kv_gemm(
    const _Float16* __restrict__ kT, const _Float16* __restrict__ svT,
    _Float16* __restrict__ kvT) {
  const int tb = blockIdx.y;
  const int m0 = (blockIdx.x >> 1) * 64;   // 4 M-blocks
  const int n0 = (blockIdx.x & 1) * 128;   // 2 N-blocks
  v8f acc[2][2] = {};
  gemm_f16_64x128(kT + (size_t)tb * NN * CC, CC,
                  svT + (size_t)tb * NN * CC, CC, m0, n0, CC, acc);
  const int wave = threadIdx.x >> 5, lane = threadIdx.x & 31;
  const int hi = lane >> 4, lo = lane & 15;
  const int wm = wave >> 2, wn = wave & 3;
  _Float16* out = kvT + (size_t)tb * NN * NN;
#pragma unroll
  for (int i = 0; i < 2; ++i)
#pragma unroll
    for (int j = 0; j < 2; ++j) {
      const int n = n0 + wn * 32 + j * 16 + lo;
#pragma unroll
      for (int r = 0; r < 8; ++r) {
        int m = m0 + wm * 32 + i * 16 + hi * 8 + r;
        out[(size_t)n * NN + m] = (_Float16)acc[i][j][r];  // transposed store
      }
    }
}

// ---------------------------------------------------------------------------
// a_pre = 0.125 * (q @ kv) : M=512, N=256, K=256. grid=(16, 64).
// ---------------------------------------------------------------------------
__global__ void __launch_bounds__(256) attn_gemm(
    const _Float16* __restrict__ sq, const _Float16* __restrict__ kvT,
    float* __restrict__ apre) {
  const int tb = blockIdx.y;
  const int m0 = (blockIdx.x >> 1) * 64;
  const int n0 = (blockIdx.x & 1) * 128;
  v8f acc[2][2] = {};
  gemm_f16_64x128(sq + (size_t)tb * CC * NN, NN,
                  kvT + (size_t)tb * NN * NN, NN, m0, n0, NN, acc);
  const int wave = threadIdx.x >> 5, lane = threadIdx.x & 31;
  const int hi = lane >> 4, lo = lane & 15;
  const int wm = wave >> 2, wn = wave & 3;
  const size_t boff = (size_t)tb * CC * NN;
#pragma unroll
  for (int i = 0; i < 2; ++i)
#pragma unroll
    for (int j = 0; j < 2; ++j) {
      const int n = n0 + wn * 32 + j * 16 + lo;
#pragma unroll
      for (int r = 0; r < 8; ++r) {
        int m = m0 + wm * 32 + i * 16 + hi * 8 + r;
        apre[boff + (size_t)m * NN + n] = acc[i][j][r] * 0.125f;
      }
    }
}

// ---------------------------------------------------------------------------
// Attention LIF: v_th = 0.5; spikes stored transposed [n][c] so they are the
// column-major B operand of the projection GEMM.
// ---------------------------------------------------------------------------
__global__ void __launch_bounds__(256) attn_lif(const float* __restrict__ ap,
                                                _Float16* __restrict__ saT) {
  const int idx = blockIdx.x * blockDim.x + threadIdx.x;
  const int b = idx / (CC * NN);
  const int c = (idx / NN) % CC;
  const int n = idx % NN;
  const size_t toff = (size_t)b * NN * CC + (size_t)n * CC + c;
  float v = 0.f;
  for (int t = 0; t < TT; ++t) {
    const size_t o = (size_t)t * SS + idx;
    v += (ap[o] - v) * 0.5f;
    float s = (v >= 0.5f) ? 1.0f : 0.0f;
    saT[(size_t)t * SS + toff] = (_Float16)s;
    v *= (1.0f - s);
  }
}

// ---------------------------------------------------------------------------
// Projection: Y = BN(p_w @ sa + p_b). M=512, N=256, K=512. grid=(16, 64).
// ---------------------------------------------------------------------------
__global__ void __launch_bounds__(256) proj_gemm(
    const _Float16* __restrict__ pwh, const _Float16* __restrict__ saT,
    const float* __restrict__ pb, const float* __restrict__ gamma,
    const float* __restrict__ beta, const float* __restrict__ mean,
    const float* __restrict__ var, float* __restrict__ Y) {
  const int tb = blockIdx.y;
  const int m0 = (blockIdx.x >> 1) * 64;
  const int n0 = (blockIdx.x & 1) * 128;
  v8f acc[2][2] = {};
  gemm_f16_64x128(pwh, CC, saT + (size_t)tb * NN * CC, CC, m0, n0, CC, acc);
  const int wave = threadIdx.x >> 5, lane = threadIdx.x & 31;
  const int hi = lane >> 4, lo = lane & 15;
  const int wm = wave >> 2, wn = wave & 3;
  const size_t boff = (size_t)tb * CC * NN;
#pragma unroll
  for (int i = 0; i < 2; ++i)
#pragma unroll
    for (int j = 0; j < 2; ++j) {
      const int n = n0 + wn * 32 + j * 16 + lo;
#pragma unroll
      for (int r = 0; r < 8; ++r) {
        int m = m0 + wm * 32 + i * 16 + hi * 8 + r;
        float inv  = gamma[m] * rsqrtf(var[m] + 1e-5f);
        float bias = pb[m] * inv + beta[m] - mean[m] * inv;
        Y[boff + (size_t)m * NN + n] = acc[i][j][r] * inv + bias;
      }
    }
}

// ---------------------------------------------------------------------------
// Final two chained LIFs; second scan continues from the first's membrane.
// ---------------------------------------------------------------------------
__global__ void __launch_bounds__(256) final_lif(const float* __restrict__ Y,
                                                 float* __restrict__ s2out) {
  const int idx = blockIdx.x * blockDim.x + threadIdx.x;
  float v = 0.f;
  float s1[TT];
  for (int t = 0; t < TT; ++t) {
    const size_t o = (size_t)t * SS + idx;
    v += (Y[o] - v) * 0.5f;
    float s = (v >= 1.0f) ? 1.0f : 0.0f;
    s1[t] = s;
    v *= (1.0f - s);
  }
  for (int t = 0; t < TT; ++t) {
    v += (s1[t] - v) * 0.5f;
    float s = (v >= 1.0f) ? 1.0f : 0.0f;
    s2out[(size_t)t * SS + idx] = s;
    v *= (1.0f - s);
  }
}

// ---------------------------------------------------------------------------
extern "C" void kernel_launch(void* const* d_in, const int* in_sizes, int n_in,
                              void* d_out, int out_size, void* d_ws,
                              size_t ws_size, hipStream_t stream) {
  // setup_inputs() dict order:
  const float* x   = (const float*)d_in[0];
  // d_in[1] = res_attn (unused by reference)
  const float* qw  = (const float*)d_in[2];
  const float* kw  = (const float*)d_in[3];
  const float* vw  = (const float*)d_in[4];
  const float* qg  = (const float*)d_in[5];
  const float* qbt = (const float*)d_in[6];
  const float* qmn = (const float*)d_in[7];
  const float* qvr = (const float*)d_in[8];
  const float* kg  = (const float*)d_in[9];
  const float* kbt = (const float*)d_in[10];
  const float* kmn = (const float*)d_in[11];
  const float* kvr = (const float*)d_in[12];
  const float* vg  = (const float*)d_in[13];
  const float* vbt = (const float*)d_in[14];
  const float* vmn = (const float*)d_in[15];
  const float* vvr = (const float*)d_in[16];
  const float* pg  = (const float*)d_in[17];
  const float* pbt = (const float*)d_in[18];
  const float* pmn = (const float*)d_in[19];
  const float* pvr = (const float*)d_in[20];
  const float* pw  = (const float*)d_in[21];
  const float* pb  = (const float*)d_in[22];
  float* out = (float*)d_out;

  const size_t E  = (size_t)TT * SS;      // 8,388,608 elements
  const size_t WW = (size_t)CC * CC;      // 262,144 weight elements
  char* ws = (char*)d_ws;
  size_t off = 0;
  _Float16* xh_T = (_Float16*)(ws + off); off += E * 2;
  _Float16* wq_h = (_Float16*)(ws + off); off += WW * 2;
  _Float16* wk_h = (_Float16*)(ws + off); off += WW * 2;
  _Float16* wv_h = (_Float16*)(ws + off); off += WW * 2;
  _Float16* wp_h = (_Float16*)(ws + off); off += WW * 2;
  float*    u_q  = (float*)(ws + off);    off += E * 4;
  float*    u_k  = (float*)(ws + off);    off += E * 4;
  float*    u_v  = (float*)(ws + off);    off += E * 4;
  _Float16* sq   = (_Float16*)(ws + off); off += E * 2;
  _Float16* skT  = (_Float16*)(ws + off); off += E * 2;
  _Float16* svT  = (_Float16*)(ws + off); off += E * 2;
  _Float16* kvT  = (_Float16*)(ws + off); off += (size_t)TBATCH * NN * NN * 2;
  // reuse (dependencies retired in stream order):
  float*    apre = u_q;
  _Float16* saT  = (_Float16*)u_k;
  float*    Y    = u_v;

  const dim3 blk(256);
  const dim3 g_gemm(16, TBATCH);          // 64x128 tiles over 512x256
  const dim3 g_kv(8, TBATCH);             // 64x128 tiles over 256x256
  const int  g_elem = SS / 256;           // 8192 blocks, LIF scans
  const int  g_xt   = (int)(E / 256);     // 32768 blocks, x transpose
  const int  g_w    = (int)(WW / 256);    // 1024 blocks, weight cvt

  cvt_w_f16<<<g_w, blk, 0, stream>>>(qw, wq_h);
  cvt_w_f16<<<g_w, blk, 0, stream>>>(kw, wk_h);
  cvt_w_f16<<<g_w, blk, 0, stream>>>(vw, wv_h);
  cvt_w_f16<<<g_w, blk, 0, stream>>>(pw, wp_h);
  transpose_x_f16<<<g_xt, blk, 0, stream>>>(x, xh_T);

  gemm_qkv_bn<<<g_gemm, blk, 0, stream>>>(wq_h, xh_T, qg, qbt, qmn, qvr, u_q);
  gemm_qkv_bn<<<g_gemm, blk, 0, stream>>>(wk_h, xh_T, kg, kbt, kmn, kvr, u_k);
  gemm_qkv_bn<<<g_gemm, blk, 0, stream>>>(wv_h, xh_T, vg, vbt, vmn, vvr, u_v);
  lif_qkv<<<g_elem, blk, 0, stream>>>(u_q, u_k, u_v, sq, skT, svT, out + E);
  kv_gemm<<<g_kv, blk, 0, stream>>>(skT, svT, kvT);
  attn_gemm<<<g_gemm, blk, 0, stream>>>(sq, kvT, apre);
  attn_lif<<<g_elem, blk, 0, stream>>>(apre, saT);
  proj_gemm<<<g_gemm, blk, 0, stream>>>(wp_h, saT, pb, pg, pbt, pmn, pvr, Y);
  final_lif<<<g_elem, blk, 0, stream>>>(Y, out);
}
